// Block_65326452572922
// MI455X (gfx1250) — matrix operand order repaired and checked by reading, not compiled, not run on hardware
//
#include <hip/hip_runtime.h>

#define CHN 64  // N_CHANNELS (layouts assume 64)

typedef __attribute__((ext_vector_type(16))) __bf16 v16bf;
typedef __attribute__((ext_vector_type(8)))  float  v8f;

__device__ __forceinline__ unsigned f2bfbits(float f) {  // RNE, result in [15:0]
  union { float f; unsigned u; } a; a.f = f;
  return (a.u + 0x7FFFu + ((a.u >> 16) & 1u)) >> 16;
}
__device__ __forceinline__ __bf16 f2bf(float f) {
  union { unsigned short s; __bf16 h; } b;
  b.s = (unsigned short)f2bfbits(f);
  return b.h;
}
__device__ __forceinline__ float bfhi2f(unsigned hi) {  // bf16 bits already in [31:16]
  union { unsigned u; float f; } a; a.u = hi; return a.f;
}

// ---------------- degree (f32 atomic counts, exact for small counts) --------
__global__ void deg_kernel(const long long* __restrict__ row,
                           float* __restrict__ deg, int E) {
  int t = blockIdx.x * blockDim.x + threadIdx.x;
  if (t < E) atomicAdd(&deg[(int)row[t]], 1.0f);
}

// ------------- norm = -dinv[row]*dinv[col]; int64 -> int32 edges ------------
__global__ void norm_kernel(const long long* __restrict__ eidx,
                            const float* __restrict__ deg,
                            int* __restrict__ row32, int* __restrict__ col32,
                            float* __restrict__ nrm, int E) {
  int t = blockIdx.x * blockDim.x + threadIdx.x;
  if (t >= E) return;
  int r = (int)eidx[t];
  int c = (int)eidx[(long long)E + t];
  row32[t] = r; col32[t] = c;
  float dr = deg[r], dc = deg[c];
  float ir = dr > 0.f ? rsqrtf(dr) : 0.f;
  float ic = dc > 0.f ? rsqrtf(dc) : 0.f;
  nrm[t] = -ir * ic;
}

// --- pack Wflat[K*64, 64] f32 -> bf16 WMMA B-fragments ----------------------
// Fragment (kc, nt): lane L holds 16 bf16:
//   elem j = Wflat[(kc*32 + (j&8)*2 + (L>>4)*8 + (j&7))*64 + (nt*16 + (L&15))]
// Output index: ((kc*4 + nt)*32 + L)*16 + j
__global__ void pack_w_kernel(const float* __restrict__ W,
                              __bf16* __restrict__ Wp, int total) {
  int t = blockIdx.x * blockDim.x + threadIdx.x;
  if (t >= total) return;
  int j    = t & 15;
  int L    = (t >> 4) & 31;
  int frag = t >> 9;
  int nt   = frag & 3;
  int kc   = frag >> 2;
  int kk = kc * 32 + (j & 8) * 2 + (L >> 4) * 8 + (j & 7);
  int n  = nt * 16 + (L & 15);
  Wp[t] = f2bf(W[(long long)kk * CHN + n]);
}

// ------------- propagate: P[row[e]] += norm[e] * x[col[e]] ------------------
// x is a bf16 slice of Tall (row stride ldx). 16 threads/edge, 4 ch each.
__global__ void scatter_kernel(const __bf16* __restrict__ x, int ldx,
                               const int* __restrict__ row,
                               const int* __restrict__ col,
                               const float* __restrict__ nrm,
                               float* __restrict__ P, int E) {
  long long t = (long long)blockIdx.x * blockDim.x + threadIdx.x;
  long long e = t >> 4;
  if (e >= E) return;
  int c = (int)(t & 15) * 4;
  int dst = row[e], src = col[e];
  float w = nrm[e];
  uint2 q = *(const uint2*)(x + (long long)src * ldx + c);
  float* o = P + (long long)dst * CHN + c;
  atomicAdd(o + 0, w * bfhi2f(q.x << 16));
  atomicAdd(o + 1, w * bfhi2f(q.x & 0xFFFF0000u));
  atomicAdd(o + 2, w * bfhi2f(q.y << 16));
  atomicAdd(o + 3, w * bfhi2f(q.y & 0xFFFF0000u));
}

// --- store f32 P into bf16 slice of Tall; HAS0: val = 2*P - T0 (Chebyshev) --
template <bool HAS0>
__global__ void store_slice4(const float4* __restrict__ P,
                             const __bf16* __restrict__ t0,
                             __bf16* __restrict__ dst, int ldt, long long n4) {
  long long t = (long long)blockIdx.x * blockDim.x + threadIdx.x;
  if (t >= n4) return;
  int row = (int)(t >> 4);
  int ch  = (int)(t & 15) * 4;
  float4 p = P[t];
  long long o = (long long)row * ldt + ch;
  if (HAS0) {
    uint2 q = *(const uint2*)(t0 + o);
    p.x = 2.f * p.x - bfhi2f(q.x << 16);
    p.y = 2.f * p.y - bfhi2f(q.x & 0xFFFF0000u);
    p.z = 2.f * p.z - bfhi2f(q.y << 16);
    p.w = 2.f * p.w - bfhi2f(q.y & 0xFFFF0000u);
  }
  uint2 w;
  w.x = f2bfbits(p.x) | (f2bfbits(p.y) << 16);
  w.y = f2bfbits(p.z) | (f2bfbits(p.w) << 16);
  *(uint2*)(dst + o) = w;
}

// ------------- OUT[N,64] = relu(A[N,KCH*32] @ W + bias) (+resid) ------------
// 8 waves/block, one 16-row stripe/wave. A is bf16 row-major (stride lda),
// loaded as raw 16B fragment halves (no conversion). Wp pre-packed fragments.
// MODE 0: write bf16 into outB (stride ldo).  MODE 1: write f32 + resid.
template <int KCH, int MODE>
__global__ __launch_bounds__(256)
void gemm_wmma(const __bf16* __restrict__ A, int lda,
               const __bf16* __restrict__ Wp,
               const float* __restrict__ bias,
               const float* __restrict__ resid,
               float* __restrict__ outF,
               __bf16* __restrict__ outB, int ldo, int nrows) {
  const int wave = threadIdx.x >> 5;
  const int lane = threadIdx.x & 31;
  const int row0 = blockIdx.x * 128 + wave * 16;
  if (row0 >= nrows) return;  // wave-uniform: EXEC all-ones for WMMA
  const int m16  = lane & 15;
  const int half = lane >> 4;

  // A fragments (16-bit A 16x32 layout): elems 0..7 = K base..+7,
  // elems 8..15 = K base+16..+23, base = kc*32 + half*8. Contiguous bf16!
  v16bf a[KCH];
  const __bf16* ar = A + (long long)(row0 + m16) * lda;
#pragma unroll
  for (int kc = 0; kc < KCH; ++kc) {
    const int base = kc * 32 + half * 8;
    union { uint4 q[2]; v16bf v; } ld;
    ld.q[0] = *(const uint4*)(ar + base);
    ld.q[1] = *(const uint4*)(ar + base + 16);
    a[kc] = ld.v;
  }

#pragma unroll
  for (int nt = 0; nt < 4; ++nt) {
    const int n = nt * 16 + m16;
    v8f c = {0.f, 0.f, 0.f, 0.f, 0.f, 0.f, 0.f, 0.f};
#pragma unroll
    for (int kc = 0; kc < KCH; ++kc) {
      const v16bf b =
          *(const v16bf*)(Wp + (((kc * 4 + nt) * 32 + lane) << 4));
      c = __builtin_amdgcn_wmma_f32_16x16x32_bf16(false, a[kc], false, b,
                                                  (short)0, c, false, false);
    }
#pragma unroll
    for (int r = 0; r < 8; ++r) {
      const int m = half * 8 + r;
      float v = c[r] + bias[n];
      v = v > 0.f ? v : 0.f;
      if (MODE == 1) {
        v += resid[(long long)(row0 + m) * CHN + n];
        outF[(long long)(row0 + m) * CHN + n] = v;
      } else {
        outB[(long long)(row0 + m) * ldo + n] = f2bf(v);
      }
    }
  }
}

// Runtime-KCH fallback (A loaded in-loop), for K_CHEB != 6.
template <int MODE>
__global__ __launch_bounds__(256)
void gemm_wmma_rt(const __bf16* __restrict__ A, int lda,
                  const __bf16* __restrict__ Wp,
                  const float* __restrict__ bias,
                  const float* __restrict__ resid,
                  float* __restrict__ outF,
                  __bf16* __restrict__ outB, int ldo, int nrows, int kch) {
  const int wave = threadIdx.x >> 5;
  const int lane = threadIdx.x & 31;
  const int row0 = blockIdx.x * 128 + wave * 16;
  if (row0 >= nrows) return;
  const int m16  = lane & 15;
  const int half = lane >> 4;
  const __bf16* ar = A + (long long)(row0 + m16) * lda;

  for (int nt = 0; nt < 4; ++nt) {
    const int n = nt * 16 + m16;
    v8f c = {0.f, 0.f, 0.f, 0.f, 0.f, 0.f, 0.f, 0.f};
    for (int kc = 0; kc < kch; ++kc) {
      const int base = kc * 32 + half * 8;
      union { uint4 q[2]; v16bf v; } ld;
      ld.q[0] = *(const uint4*)(ar + base);
      ld.q[1] = *(const uint4*)(ar + base + 16);
      const v16bf b =
          *(const v16bf*)(Wp + (((kc * 4 + nt) * 32 + lane) << 4));
      c = __builtin_amdgcn_wmma_f32_16x16x32_bf16(false, ld.v, false, b,
                                                  (short)0, c, false, false);
    }
    for (int r = 0; r < 8; ++r) {
      const int m = half * 8 + r;
      float v = c[r] + bias[n];
      v = v > 0.f ? v : 0.f;
      if (MODE == 1) {
        v += resid[(long long)(row0 + m) * CHN + n];
        outF[(long long)(row0 + m) * CHN + n] = v;
      } else {
        outB[(long long)(row0 + m) * ldo + n] = f2bf(v);
      }
    }
  }
}

extern "C" void kernel_launch(void* const* d_in, const int* in_sizes, int n_in,
                              void* d_out, int out_size, void* d_ws,
                              size_t ws_size, hipStream_t stream) {
  const float* pos = (const float*)d_in[0];
  const long long* eidx = (const long long*)d_in[1];
  const float* Wl[3] = {(const float*)d_in[2], (const float*)d_in[4],
                        (const float*)d_in[6]};
  const float* bl[3] = {(const float*)d_in[3], (const float*)d_in[5],
                        (const float*)d_in[7]};

  const int N = in_sizes[0] / CHN;          // 100000 (multiple of 16)
  const int E = in_sizes[1] / 2;            // 1200000
  const int K = in_sizes[2] / (CHN * CHN);  // 6
  const int KTOT = K * CHN;                 // 384
  const int KCHn = KTOT / 32;               // 12
  const long long NC = (long long)N * CHN;
  const long long NT = (long long)N * KTOT;

  // ---- workspace carve-out (256B aligned) ----
  char* ws = (char*)d_ws;
  size_t off = 0;
  auto alloc = [&](size_t bytes) -> char* {
    off = (off + 255) & ~(size_t)255;
    char* p = ws + off;
    off += bytes;
    return p;
  };
  float*  deg   = (float*)alloc(sizeof(float) * N);
  int*    row32 = (int*)alloc(sizeof(int) * E);
  int*    col32 = (int*)alloc(sizeof(int) * E);
  float*  nrm   = (float*)alloc(sizeof(float) * E);
  __bf16* Wp[3];
  Wp[0] = (__bf16*)alloc(sizeof(__bf16) * KTOT * CHN);
  Wp[1] = (__bf16*)alloc(sizeof(__bf16) * KTOT * CHN);
  Wp[2] = (__bf16*)alloc(sizeof(__bf16) * KTOT * CHN);
  __bf16* TA = (__bf16*)alloc(sizeof(__bf16) * NT);  // Tall ping
  __bf16* TB = (__bf16*)alloc(sizeof(__bf16) * NT);  // Tall pong
  float*  P  = (float*)alloc(sizeof(float) * NC);    // f32 scatter accumulator
  (void)ws_size; (void)n_in; (void)out_size;

  // ---- graph normalization + weight packing ----
  hipMemsetAsync(deg, 0, sizeof(float) * N, stream);
  deg_kernel<<<(E + 255) / 256, 256, 0, stream>>>(eidx, deg, E);
  norm_kernel<<<(E + 255) / 256, 256, 0, stream>>>(eidx, deg, row32, col32,
                                                   nrm, E);
  for (int l = 0; l < 3; ++l)
    pack_w_kernel<<<(KTOT * CHN + 255) / 256, 256, 0, stream>>>(
        Wl[l], Wp[l], KTOT * CHN);

  const int ggrid = (N + 127) / 128;
  const long long n4 = NC / 4;
  const int g4 = (int)((n4 + 255) / 256);
  const long long stot = (long long)E * 16;
  const int gs = (int)((stot + 255) / 256);

  auto run_layer = [&](__bf16* Tall, const __bf16* Wpk, const float* b,
                       __bf16* outB, float* outF, const float* resid) {
    // Build Chebyshev slices 1..K-1 (slice 0 already holds this layer's x).
    for (int k = 1; k < K; ++k) {
      hipMemsetAsync(P, 0, sizeof(float) * NC, stream);
      scatter_kernel<<<gs, 256, 0, stream>>>(
          Tall + (long long)(k - 1) * CHN, KTOT, row32, col32, nrm, P, E);
      if (k == 1)
        store_slice4<false><<<g4, 256, 0, stream>>>(
            (const float4*)P, nullptr, Tall + CHN, KTOT, n4);
      else
        store_slice4<true><<<g4, 256, 0, stream>>>(
            (const float4*)P, Tall + (long long)(k - 2) * CHN,
            Tall + (long long)k * CHN, KTOT, n4);
    }
    // One fused GEMM over K = KTOT.
    if (outF) {
      if (KCHn == 12)
        gemm_wmma<12, 1><<<ggrid, 256, 0, stream>>>(
            Tall, KTOT, Wpk, b, resid, outF, nullptr, 0, N);
      else
        gemm_wmma_rt<1><<<ggrid, 256, 0, stream>>>(
            Tall, KTOT, Wpk, b, resid, outF, nullptr, 0, N, KCHn);
    } else {
      if (KCHn == 12)
        gemm_wmma<12, 0><<<ggrid, 256, 0, stream>>>(
            Tall, KTOT, Wpk, b, nullptr, nullptr, outB, KTOT, N);
      else
        gemm_wmma_rt<0><<<ggrid, 256, 0, stream>>>(
            Tall, KTOT, Wpk, b, nullptr, nullptr, outB, KTOT, N, KCHn);
    }
  };

  // Layer 1 input: slice 0 of TA = bf16(pos)
  store_slice4<false><<<g4, 256, 0, stream>>>((const float4*)pos, nullptr, TA,
                                              KTOT, n4);
  run_layer(TA, Wp[0], bl[0], TB, nullptr, nullptr);  // h1 -> TB slice 0
  run_layer(TB, Wp[1], bl[1], TA, nullptr, nullptr);  // h2 -> TA slice 0
  run_layer(TA, Wp[2], bl[2], nullptr, (float*)d_out, pos);  // f32 + residual
}
